// DeformConv_9234179687513
// MI455X (gfx1250) — compile-verified, matrix operand order checked by reading, not run on hardware
//
#include <hip/hip_runtime.h>
#include <hip/hip_bf16.h>

typedef __attribute__((ext_vector_type(16))) __bf16 v16bf;
typedef __attribute__((ext_vector_type(8)))  __bf16 v8bf;
typedef __attribute__((ext_vector_type(8)))  float  v8f;

#define CIN    256
#define COUT   256
#define Hh     64
#define Ww     64
#define PIX    4096          // Ho*Wo
#define KTAPS  9
#define NT     64            // pixels per block (N tile)
#define NSTEPS 72            // 9 taps * (256/32) channel blocks
#define CSTRIDE 56           // colsT row stride in bf16 elems (112 B, 16B aligned)

__device__ __forceinline__ unsigned bf16b(float f) {
  unsigned u = __float_as_uint(f);
  u += 0x7fffu + ((u >> 16) & 1u);        // round-to-nearest-even
  return u >> 16;
}
__device__ __forceinline__ __bf16 f2bf(float f) {
  unsigned short h = (unsigned short)bf16b(f);
  __bf16 r;
  __builtin_memcpy(&r, &h, 2);
  return r;
}

// ---------------------------------------------------------------------------
// Prepack weight[Cout, Cin, 3, 3] (f32) -> bf16 A-fragments in WMMA lane order.
// K order: k_global = tap*256 + channel; step s = tap*8 + cb covers 32 channels.
// wf[((s*16 + mt)*32 + lane)*16 + e], e->k per ISA 16-bit A 16x32 layout.
// ---------------------------------------------------------------------------
__global__ __launch_bounds__(256) void prepack_w(const float* __restrict__ w,
                                                 __bf16* __restrict__ wf) {
  int gid   = blockIdx.x * 256 + threadIdx.x;   // < 72*16*32*16 = 589824
  int e     = gid & 15;
  int lane  = (gid >> 4) & 31;
  int mt    = (gid >> 9) & 15;
  int s     = gid >> 13;                        // 0..71
  int t     = s >> 3;                           // tap
  int cb    = s & 7;                            // channel block
  int khalf = lane >> 4;
  int kl    = (e < 8) ? (khalf * 8 + e) : (16 + khalf * 8 + (e - 8));
  int c     = cb * 32 + kl;
  int m     = mt * 16 + (lane & 15);
  wf[gid] = f2bf(w[((size_t)m * CIN + c) * (KTAPS) + t]);
}

// ---------------------------------------------------------------------------
// Fused deformable conv: bilinear im2col into LDS (bf16) + WMMA GEMM.
// One block: full M=256 x N=64 pixel tile of one batch. 8 waves, wave w owns
// rows [32w, 32w+32). 72 K-steps of 32.
// ---------------------------------------------------------------------------
__global__ __launch_bounds__(256) void deform_wmma(const float* __restrict__ x,
                                                   const float* __restrict__ off,
                                                   const __bf16* __restrict__ wfrag,
                                                   float* __restrict__ out) {
  __shared__ alignas(16) int    p_off[KTAPS * NT * 4];
  __shared__ alignas(16) float  p_wgt[KTAPS * NT * 4];
  __shared__ alignas(16) __bf16 colsT[NT * CSTRIDE];

  const int b    = blockIdx.x >> 6;
  const int n0   = (blockIdx.x & 63) * NT;
  const int tid  = threadIdx.x;
  const int lane = tid & 31;
  const int wave = tid >> 5;
  const int lrow = lane & 15;
  const int khalf = lane >> 4;

  // ---- bilinear params: per (tap, pixel): 4 clamped offsets + 4 weights ----
  const float* offb = off + (size_t)b * (2 * KTAPS) * PIX;
  for (int idx = tid; idx < KTAPS * NT; idx += 256) {
    int t = idx >> 6;
    int n = idx & 63;
    int p = n0 + n;                 // global pixel
    int ho = p >> 6, wo = p & 63;
    float oy = offb[(2 * t) * PIX + p];
    float ox = offb[(2 * t + 1) * PIX + p];
    float py = (float)(ho - 1 + (t / 3)) + oy;
    float px = (float)(wo - 1 + (t % 3)) + ox;
    float y0f = floorf(py), x0f = floorf(px);
    float wy = py - y0f, wx = px - x0f;
    int y0 = (int)y0f, x0 = (int)x0f;
    int y1 = y0 + 1, x1 = x0 + 1;
    float vy0 = (y0 >= 0 && y0 < Hh) ? 1.f : 0.f;
    float vy1 = (y1 >= 0 && y1 < Hh) ? 1.f : 0.f;
    float vx0 = (x0 >= 0 && x0 < Ww) ? 1.f : 0.f;
    float vx1 = (x1 >= 0 && x1 < Ww) ? 1.f : 0.f;
    int y0c = min(max(y0, 0), Hh - 1), y1c = min(max(y1, 0), Hh - 1);
    int x0c = min(max(x0, 0), Ww - 1), x1c = min(max(x1, 0), Ww - 1);
    *(int4*)&p_off[idx * 4] =
        make_int4(y0c * Ww + x0c, y0c * Ww + x1c, y1c * Ww + x0c, y1c * Ww + x1c);
    *(float4*)&p_wgt[idx * 4] =
        make_float4((1.f - wy) * (1.f - wx) * vy0 * vx0,
                    (1.f - wy) * wx * vy0 * vx1,
                    wy * (1.f - wx) * vy1 * vx0,
                    wy * wx * vy1 * vx1);
  }

  v8f acc[2][4];
  #pragma unroll
  for (int i = 0; i < 2; ++i)
    #pragma unroll
    for (int j = 0; j < 4; ++j)
      acc[i][j] = (v8f)0.f;

  const float* xb  = x + (size_t)b * CIN * PIX;
  const int    gn  = tid & 63;           // pixel within tile this thread gathers
  const int    gkb = (tid >> 6) << 3;    // k sub-block 0/8/16/24

  __syncthreads();

  for (int t = 0; t < KTAPS; ++t) {
    const int4   o4 = *(const int4*)&p_off[(t * 64 + gn) * 4];
    const float4 w4 = *(const float4*)&p_wgt[(t * 64 + gn) * 4];
    for (int cb = 0; cb < 8; ++cb) {
      const int s = t * 8 + cb;
      __syncthreads();                                   // colsT consumed
      // ---- gather 8 channels for (pixel gn, tap t), round to bf16 ----
      const float* xc0 = xb + (size_t)(cb * 32 + gkb) * PIX;
      unsigned pk[4];
      #pragma unroll
      for (int j = 0; j < 8; ++j) {
        const float* xc = xc0 + (size_t)j * PIX;
        float v = w4.x * xc[o4.x] + w4.y * xc[o4.y] +
                  w4.z * xc[o4.z] + w4.w * xc[o4.w];
        unsigned hb = bf16b(v);
        if (j & 1) pk[j >> 1] |= hb << 16; else pk[j >> 1] = hb;
      }
      *(uint4*)&colsT[gn * CSTRIDE + gkb] = make_uint4(pk[0], pk[1], pk[2], pk[3]);
      __syncthreads();                                   // colsT visible

      // ---- A fragments: prepacked, coalesced 32B loads from L2 ----
      v16bf afrag[2];
      #pragma unroll
      for (int i = 0; i < 2; ++i) {
        int mt = wave * 2 + i;
        afrag[i] = *(const v16bf*)(wfrag + ((size_t)(s * 16 + mt) * 32 + lane) * 16);
      }
      // ---- B fragments: two aligned 16B LDS reads + concat ----
      v16bf bfrag[4];
      #pragma unroll
      for (int j = 0; j < 4; ++j) {
        const __bf16* bp = &colsT[(j * 16 + lrow) * CSTRIDE + khalf * 8];
        v8bf lo = *(const v8bf*)bp;            // K = khalf*8 + 0..7
        v8bf hi = *(const v8bf*)(bp + 16);     // K = 16 + khalf*8 + 0..7
        bfrag[j] = __builtin_shufflevector(lo, hi, 0, 1, 2, 3, 4, 5, 6, 7,
                                           8, 9, 10, 11, 12, 13, 14, 15);
      }
      // ---- 8 WMMAs: 32x64 tile advance by K=32 ----
      #pragma unroll
      for (int i = 0; i < 2; ++i)
        #pragma unroll
        for (int j = 0; j < 4; ++j)
          acc[i][j] = __builtin_amdgcn_wmma_f32_16x16x32_bf16(
              false, afrag[i], false, bfrag[j], (short)0, acc[i][j], false, false);
    }
  }

  // ---- epilogue: f32 C/D layout -> out[b, m, p] ----
  float* outb = out + (size_t)b * COUT * PIX;
  const int mofs = khalf * 8;
  #pragma unroll
  for (int i = 0; i < 2; ++i) {
    #pragma unroll
    for (int j = 0; j < 4; ++j) {
      int col   = n0 + j * 16 + lrow;
      int mbase = wave * 32 + i * 16 + mofs;
      #pragma unroll
      for (int r = 0; r < 8; ++r)
        outb[(size_t)(mbase + r) * PIX + col] = acc[i][j][r];
    }
  }
}

extern "C" void kernel_launch(void* const* d_in, const int* in_sizes, int n_in,
                              void* d_out, int out_size, void* d_ws, size_t ws_size,
                              hipStream_t stream) {
  const float* x      = (const float*)d_in[0];
  const float* offset = (const float*)d_in[1];
  const float* weight = (const float*)d_in[2];
  float*       out    = (float*)d_out;
  __bf16*      wfrag  = (__bf16*)d_ws;   // 72*16*32*16 bf16 = 1.18 MB

  prepack_w<<<2304, 256, 0, stream>>>(weight, wfrag);
  deform_wmma<<<8 * 64, 256, 0, stream>>>(x, offset, wfrag, out);
}
